// MS_transformer_91293824844393
// MI455X (gfx1250) — matrix operand order repaired and checked by reading, not compiled, not run on hardware
//
#include <hip/hip_runtime.h>

typedef __attribute__((ext_vector_type(16))) _Float16 v16h;
typedef __attribute__((ext_vector_type(4)))  _Float16 v4h;
typedef __attribute__((ext_vector_type(8)))  float    v8f;

#define Bn    16
#define Cn    128
#define Ln    3000
#define CINn  128
#define Kt    5
#define WINn  25
#define NWn   120
#define NHn   8
#define SHIFTn 12
#define Mtok  (Bn * Ln)     /* 48000 tokens */
#define EPSn  1e-5f

/* ------------------------------------------------------------------ */
/* WMMA fragment helpers (wave32, 16x16x32 f16)                        */
/*  A (16x32):  lane m=l&15, hi=l>>4; halves at K = hi*8+w, 16+hi*8+w  */
/*  B (32x16):  lane n=l&15, hi=l>>4; halves at K = hi*8+w, 16+hi*8+w  */
/* ------------------------------------------------------------------ */
union Frag { v16h v; uint4 q[2]; _Float16 h[16]; };

__device__ __forceinline__ v16h frag_load_a(const _Float16* p) {
  Frag f; f.q[0] = *(const uint4*)(p); f.q[1] = *(const uint4*)(p + 16); return f.v;
}
__device__ __forceinline__ v16h frag_load_b16(const _Float16* p) {
  Frag f; f.q[0] = *(const uint4*)(p); f.q[1] = *(const uint4*)(p + 16); return f.v;
}
__device__ __forceinline__ v16h frag_zero() {
  Frag f; f.q[0] = make_uint4(0u,0u,0u,0u); f.q[1] = make_uint4(0u,0u,0u,0u); return f.v;
}
__device__ __forceinline__ v8f wmma16(v16h a, v16h b, v8f c) {
  return __builtin_amdgcn_wmma_f32_16x16x32_f16(false, a, false, b, (short)0, c, false, false);
}

/* ------------------------------------------------------------------ */
/* Dense GEMM: out[M,N] = A[M,K](f16) @ W[N,K]^T(f16) + bias           */
/* Register-blocked: each wave computes a 32(M) x 64(N) tile:          */
/*   per k-step: 2 A frags + 4 B frags -> 8 WMMAs                      */
/* ------------------------------------------------------------------ */
#define EPI_F16  0   /* out16[m*N+n] = v                 */
#define EPI_GELU 1   /* out16[m*N+n] = gelu(v)           */
#define EPI_RES  2   /* resid[roll(m,shift)*N+n] += v    */

__global__ __launch_bounds__(128) void gemm_f16_kernel(
    const _Float16* __restrict__ A, const _Float16* __restrict__ W,
    const float* __restrict__ bias, _Float16* __restrict__ out16,
    float* __restrict__ resid, int N, int K, int epi, int shift)
{
  const int wave = threadIdx.x >> 5, lane = threadIdx.x & 31;
  const int hi = lane >> 4, ml = lane & 15;
  const int m32 = (blockIdx.y * 4 + wave) * 32;
  const int n64 = blockIdx.x * 64;

  const _Float16* arow0 = A + (size_t)(m32 + ml) * K + hi * 8;
  const _Float16* arow1 = A + (size_t)(m32 + 16 + ml) * K + hi * 8;
  const _Float16* brow  = W + (size_t)(n64 + ml) * K + hi * 8;
  const size_t bstride = (size_t)16 * K;

  v8f acc[2][4];
#pragma unroll
  for (int mi = 0; mi < 2; ++mi)
#pragma unroll
    for (int nt = 0; nt < 4; ++nt)
      acc[mi][nt] = (v8f){0.f,0.f,0.f,0.f,0.f,0.f,0.f,0.f};

  for (int k0 = 0; k0 < K; k0 += 32) {
    __builtin_prefetch(arow0 + k0 + 64, 0, 0);
    __builtin_prefetch(brow + k0 + 64, 0, 0);
    v16h a0 = frag_load_a(arow0 + k0);
    v16h a1 = frag_load_a(arow1 + k0);
#pragma unroll
    for (int nt = 0; nt < 4; ++nt) {
      v16h bf = frag_load_b16(brow + bstride * nt + k0);
      acc[0][nt] = wmma16(a0, bf, acc[0][nt]);
      acc[1][nt] = wmma16(a1, bf, acc[1][nt]);
    }
  }

  if (epi == EPI_RES) {
#pragma unroll
    for (int mi = 0; mi < 2; ++mi)
#pragma unroll
      for (int nt = 0; nt < 4; ++nt) {
        const int n = n64 + nt * 16 + ml;
        const float bs = bias[n];
#pragma unroll
        for (int r = 0; r < 8; ++r) {
          int m = m32 + mi * 16 + r + hi * 8;
          int b = m / Ln, l = m - b * Ln;
          size_t dst = (size_t)(b * Ln + (l + shift) % Ln) * N + n;
          resid[dst] += acc[mi][nt][r] + bs;
        }
      }
  } else if (epi == EPI_GELU) {
#pragma unroll
    for (int mi = 0; mi < 2; ++mi)
#pragma unroll
      for (int nt = 0; nt < 4; ++nt) {
        const int n = n64 + nt * 16 + ml;
        const float bs = bias[n];
#pragma unroll
        for (int r = 0; r < 8; ++r) {
          float v = acc[mi][nt][r] + bs;
          v = 0.5f * v * (1.f + erff(v * 0.70710678118f));
          out16[(size_t)(m32 + mi * 16 + r + hi * 8) * N + n] = (_Float16)v;
        }
      }
  } else {
#pragma unroll
    for (int mi = 0; mi < 2; ++mi)
#pragma unroll
      for (int nt = 0; nt < 4; ++nt) {
        const int n = n64 + nt * 16 + ml;
        const float bs = bias[n];
#pragma unroll
        for (int r = 0; r < 8; ++r)
          out16[(size_t)(m32 + mi * 16 + r + hi * 8) * N + n] =
              (_Float16)(acc[mi][nt][r] + bs);
      }
  }
}

/* ------------------------------------------------------------------ */
/* Conv(K=5) as 5-tap WMMA GEMM, register-blocked 32(M) x 64(N).       */
/* X:[B*L,Cin] f16  Wp:[5][Cout][Cin] f16                              */
/* epilogue: bias -> relu -> BN(g/sqrt(1+eps), b)                      */
/* ------------------------------------------------------------------ */
#define CEPI_F32_BLC 0   /* out32[tok*Cout+n]            */
#define CEPI_F16_BLC 1   /* out16[tok*Cout+n]            */
#define CEPI_F32_BCL 2   /* out32[(b*Cout+n)*L + l]      */

__global__ __launch_bounds__(128) void conv_gemm_kernel(
    const _Float16* __restrict__ X, const _Float16* __restrict__ Wp,
    const float* __restrict__ bias, const float* __restrict__ bng,
    const float* __restrict__ bnb, _Float16* __restrict__ out16,
    float* __restrict__ out32, int Cin, int Cout, int dil, int epi)
{
  const int wave = threadIdx.x >> 5, lane = threadIdx.x & 31;
  const int hi = lane >> 4, ml = lane & 15;
  const int m32 = (blockIdx.y * 4 + wave) * 32;
  const int n64 = blockIdx.x * 64;

  /* this lane's two A rows (tokens m32+ml and m32+16+ml) */
  const int m0 = m32 + ml, m1 = m32 + 16 + ml;
  const int b0 = m0 / Ln, l0 = m0 - b0 * Ln;
  const int b1 = m1 / Ln, l1 = m1 - b1 * Ln;
  const size_t bstride = (size_t)16 * Cin;

  v8f acc[2][4];
#pragma unroll
  for (int mi = 0; mi < 2; ++mi)
#pragma unroll
    for (int nt = 0; nt < 4; ++nt)
      acc[mi][nt] = (v8f){0.f,0.f,0.f,0.f,0.f,0.f,0.f,0.f};

  for (int t = 0; t < Kt; ++t) {
    const int lp0 = l0 + (t - 2) * dil, lp1 = l1 + (t - 2) * dil;
    const bool v0 = (lp0 >= 0) && (lp0 < Ln);
    const bool v1 = (lp1 >= 0) && (lp1 < Ln);
    const _Float16* arow0 = X + (size_t)(b0 * Ln + (v0 ? lp0 : 0)) * Cin + hi * 8;
    const _Float16* arow1 = X + (size_t)(b1 * Ln + (v1 ? lp1 : 0)) * Cin + hi * 8;
    const _Float16* brow  = Wp + (size_t)(t * Cout + n64 + ml) * Cin + hi * 8;
    for (int k0 = 0; k0 < Cin; k0 += 32) {
      v16h a0 = v0 ? frag_load_a(arow0 + k0) : frag_zero();
      v16h a1 = v1 ? frag_load_a(arow1 + k0) : frag_zero();
#pragma unroll
      for (int nt = 0; nt < 4; ++nt) {
        v16h bf = frag_load_b16(brow + bstride * nt + k0);
        acc[0][nt] = wmma16(a0, bf, acc[0][nt]);
        acc[1][nt] = wmma16(a1, bf, acc[1][nt]);
      }
    }
  }

#pragma unroll
  for (int mi = 0; mi < 2; ++mi)
#pragma unroll
    for (int nt = 0; nt < 4; ++nt) {
      const int n = n64 + nt * 16 + ml;
      const float s = bng[n] * rsqrtf(1.f + EPSn);
      const float bb = bnb[n], bi = bias[n];
#pragma unroll
      for (int r = 0; r < 8; ++r) {
        int mr = m32 + mi * 16 + r + hi * 8;
        float v = fmaxf(acc[mi][nt][r] + bi, 0.f) * s + bb;
        if (epi == CEPI_F32_BLC)      out32[(size_t)mr * Cout + n] = v;
        else if (epi == CEPI_F16_BLC) out16[(size_t)mr * Cout + n] = (_Float16)v;
        else { int br = mr / Ln, lr = mr - br * Ln;
               out32[((size_t)br * Cout + n) * Ln + lr] = v; }
      }
    }
}

/* ------------------------------------------------------------------ */
/* LayerNorm over C=128, optional forward roll folded into the gather  */
/* one wave per token, float4 per lane + shfl_xor reduction            */
/* ------------------------------------------------------------------ */
__global__ __launch_bounds__(256) void ln_kernel(
    const float* __restrict__ X, const float* __restrict__ g,
    const float* __restrict__ be, _Float16* __restrict__ H, int shift)
{
  const int wid = (blockIdx.x * 256 + threadIdx.x) >> 5;
  const int lane = threadIdx.x & 31;
  const int b = wid / Ln, l = wid - b * Ln;
  const float4 x = *(const float4*)(X + (size_t)(b * Ln + (l + shift) % Ln) * Cn + lane * 4);
  float s  = x.x + x.y + x.z + x.w;
  float s2 = x.x*x.x + x.y*x.y + x.z*x.z + x.w*x.w;
#pragma unroll
  for (int off = 16; off > 0; off >>= 1) {
    s  += __shfl_xor(s,  off, 32);
    s2 += __shfl_xor(s2, off, 32);
  }
  const float mean = s * (1.f / Cn);
  const float inv  = rsqrtf(s2 * (1.f / Cn) - mean * mean + EPSn);
  const float4 gv = *(const float4*)(g  + lane * 4);
  const float4 bv = *(const float4*)(be + lane * 4);
  v4h o = { (_Float16)((x.x - mean) * inv * gv.x + bv.x),
            (_Float16)((x.y - mean) * inv * gv.y + bv.y),
            (_Float16)((x.z - mean) * inv * gv.z + bv.z),
            (_Float16)((x.w - mean) * inv * gv.w + bv.w) };
  *(v4h*)(H + (size_t)wid * Cn + lane * 4) = o;
}

/* ------------------------------------------------------------------ */
/* Windowed attention: one wave per (window, head). WIN=25 padded to   */
/* 32; scores & PV via WMMA (K padded), softmax + RPB + mask in VALU.  */
/* ------------------------------------------------------------------ */
__global__ __launch_bounds__(128) void attn_kernel(
    const _Float16* __restrict__ QKV, const float* __restrict__ rp,
    _Float16* __restrict__ O, int sh)
{
  __shared__ __align__(16) float    sc[4][32][32];
  __shared__ __align__(16) _Float16 at[4][32][32];
  __shared__ __align__(16) _Float16 vt[4][16][32];

  const int wave = threadIdx.x >> 5, lane = threadIdx.x & 31;
  const int hi = lane >> 4, ml = lane & 15;
  const int u = blockIdx.x * 4 + wave;          /* B*NW*NH units */
  const int head = u & (NHn - 1);
  const int wg = u / NHn;
  const int b = wg / NWn, nw = wg - b * NWn;
  const size_t tok0 = (size_t)b * Ln + nw * WINn;
  const size_t base = tok0 * 384;

  /* stage V^T[d][key] into LDS (keys >= 25 zero-padded) */
  for (int idx = lane; idx < 16 * 32; idx += 32) {
    int d = idx >> 5, key = idx & 31;
    _Float16 val = (_Float16)0.f;
    if (key < WINn) val = QKV[base + (size_t)key * 384 + 256 + head * 16 + d];
    vt[wave][d][key] = val;
  }

  /* scores = (q * 0.25) @ k^T, K padded 16->32 */
  for (int mt = 0; mt < 2; ++mt) {
    const int q = mt * 16 + ml;
    Frag fa;
    fa.q[0] = (q < WINn) ? *(const uint4*)(QKV + base + (size_t)q * 384 + head * 16 + hi * 8)
                         : make_uint4(0u,0u,0u,0u);
    fa.q[1] = make_uint4(0u,0u,0u,0u);
    for (int nt = 0; nt < 2; ++nt) {
      const int key = nt * 16 + ml;
      Frag fb;
      if (key < WINn) {
        const _Float16* kp = QKV + base + (size_t)key * 384 + 128 + head * 16 + hi * 8;
        fb.q[0] = *(const uint4*)(kp);
        fb.q[1] = make_uint4(0u,0u,0u,0u);
      } else {
        fb.q[0] = make_uint4(0u,0u,0u,0u);
        fb.q[1] = make_uint4(0u,0u,0u,0u);
      }
      v8f acc = {0.f,0.f,0.f,0.f,0.f,0.f,0.f,0.f};
      acc = wmma16(fa.v, fb.v, acc);
#pragma unroll
      for (int r = 0; r < 8; ++r)
        sc[wave][mt * 16 + r + hi * 8][nt * 16 + ml] = acc[r] * 0.25f;
    }
  }
  __syncthreads();

  /* softmax rows: lane = query row, RPB + Swin shift mask */
  {
    const int row = lane;
    if (row < WINn) {
      float mx = -1e30f;
      for (int kc = 0; kc < WINn; ++kc) {
        float bias = rp[(row - kc + WINn - 1) * NHn + head];
        if (sh && nw == NWn - 1 &&
            ((row < WINn - SHIFTn) != (kc < WINn - SHIFTn))) bias -= 100.f;
        float v = sc[wave][row][kc] + bias;
        sc[wave][row][kc] = v;
        mx = fmaxf(mx, v);
      }
      float sum = 0.f;
      for (int kc = 0; kc < WINn; ++kc) {
        float e = __expf(sc[wave][row][kc] - mx);
        sum += e;
        sc[wave][row][kc] = e;
      }
      float inv = 1.f / sum;
      for (int kc = 0; kc < 32; ++kc)
        at[wave][row][kc] = (kc < WINn) ? (_Float16)(sc[wave][row][kc] * inv)
                                        : (_Float16)0.f;
    } else {
      for (int kc = 0; kc < 32; ++kc) at[wave][row][kc] = (_Float16)0.f;
    }
  }
  __syncthreads();

  /* o = attn @ v  (K = 32 keys incl. zero pad) */
  for (int mt = 0; mt < 2; ++mt) {
    const int row = mt * 16 + ml;
    Frag fa, fb;
    fa.q[0] = *(const uint4*)&at[wave][row][hi * 8];
    fa.q[1] = *(const uint4*)&at[wave][row][16 + hi * 8];
    fb.q[0] = *(const uint4*)&vt[wave][ml][hi * 8];
    fb.q[1] = *(const uint4*)&vt[wave][ml][16 + hi * 8];
    v8f acc = {0.f,0.f,0.f,0.f,0.f,0.f,0.f,0.f};
    acc = wmma16(fa.v, fb.v, acc);
#pragma unroll
    for (int r = 0; r < 8; ++r) {
      int qr = mt * 16 + r + hi * 8;
      if (qr < WINn)
        O[(tok0 + qr) * Cn + head * 16 + ml] = (_Float16)acc[r];
    }
  }
}

/* ------------------------------------------------------------------ */
/* small utility kernels                                               */
/* ------------------------------------------------------------------ */
__global__ void cvt_kernel(const float* __restrict__ in, _Float16* __restrict__ out, int n) {
  int i = blockIdx.x * blockDim.x + threadIdx.x;
  if (i < n) out[i] = (_Float16)in[i];
}
/* in[g][co][ci][Kt] -> out[g][k][co][ci] fp16 */
__global__ void packw_kernel(const float* __restrict__ in, _Float16* __restrict__ out,
                             int G, int Co, int Ci) {
  size_t i = (size_t)blockIdx.x * blockDim.x + threadIdx.x;
  size_t tot = (size_t)G * Co * Ci * Kt;
  if (i >= tot) return;
  int k = (int)(i % Kt); size_t t = i / Kt;
  int ci = (int)(t % Ci); t /= Ci;
  int co = (int)(t % Co); int g = (int)(t / Co);
  out[(((size_t)g * Kt + k) * Co + co) * Ci + ci] = (_Float16)in[i];
}
/* x[b][cin][l] fp32 -> xT[b*L+l][cin] fp16 */
__global__ void xpose_kernel(const float* __restrict__ X, _Float16* __restrict__ XT) {
  size_t i = (size_t)blockIdx.x * blockDim.x + threadIdx.x;
  if (i >= (size_t)Bn * CINn * Ln) return;
  int l = (int)(i % Ln); size_t t = i / Ln;
  int cin = (int)(t % CINn); int b = (int)(t / CINn);
  XT[((size_t)b * Ln + l) * CINn + cin] = (_Float16)X[i];
}
/* branch bn_n + scatter into concat buffer xcT[tok][4C] fp16 */
__global__ void branch_out_kernel(const float* __restrict__ Xb, const float* __restrict__ g,
                                  const float* __restrict__ be, _Float16* __restrict__ XC, int br) {
  size_t i = (size_t)blockIdx.x * blockDim.x + threadIdx.x;
  if (i >= (size_t)Mtok * Cn) return;
  int c = (int)(i % Cn); size_t tok = i / Cn;
  float s = g[c] * rsqrtf(1.f + EPSn);
  XC[tok * (4 * Cn) + (size_t)br * Cn + c] = (_Float16)(Xb[i] * s + be[c]);
}
__global__ void mean_kernel(const _Float16* __restrict__ XC, float* __restrict__ M) {
  int i = blockIdx.x * blockDim.x + threadIdx.x;
  if (i >= Bn * 512) return;
  int ch = i % 512, b = i / 512;
  const _Float16* p = XC + (size_t)b * Ln * 512 + ch;
  float s = 0.f;
  for (int l = 0; l < Ln; ++l) s += (float)p[(size_t)l * 512];
  M[i] = s * (1.f / Ln);
}
__global__ void se1_kernel(const float* __restrict__ M, const float* __restrict__ W1,
                           float* __restrict__ T1) {
  int i = blockIdx.x * blockDim.x + threadIdx.x;
  if (i >= Bn * 256) return;
  int o = i % 256, b = i / 256;
  float s = 0.f;
  for (int k = 0; k < 512; ++k) s += M[b * 512 + k] * W1[o * 512 + k];
  T1[i] = fmaxf(s, 0.f);
}
__global__ void se2_kernel(const float* __restrict__ T1, const float* __restrict__ W2,
                           float* __restrict__ G) {
  int i = blockIdx.x * blockDim.x + threadIdx.x;
  if (i >= Bn * 512) return;
  int o = i % 512, b = i / 512;
  float s = 0.f;
  for (int k = 0; k < 256; ++k) s += T1[b * 256 + k] * W2[o * 256 + k];
  G[i] = 1.f / (1.f + __expf(-s));
}
__global__ void gate_kernel(_Float16* __restrict__ XC, const float* __restrict__ G) {
  size_t i = (size_t)blockIdx.x * blockDim.x + threadIdx.x;
  if (i >= (size_t)Mtok * 512) return;
  int ch = (int)(i % 512); size_t tok = i / 512;
  int b = (int)(tok / Ln);
  XC[i] = (_Float16)((float)XC[i] * G[b * 512 + ch]);
}

/* ------------------------------------------------------------------ */
extern "C" void kernel_launch(void* const* d_in, const int* in_sizes, int n_in,
                              void* d_out, int out_size, void* d_ws, size_t ws_size,
                              hipStream_t stream) {
  const float* x       = (const float*)d_in[0];
  const float* conv_w  = (const float*)d_in[1];
  const float* conv_b  = (const float*)d_in[2];
  const float* bn_a_g  = (const float*)d_in[3];
  const float* bn_a_b  = (const float*)d_in[4];
  const float* ln1_g   = (const float*)d_in[5];
  const float* ln1_b   = (const float*)d_in[6];
  const float* qkv_w   = (const float*)d_in[7];
  const float* qkv_b   = (const float*)d_in[8];
  const float* proj_w  = (const float*)d_in[9];
  const float* proj_b  = (const float*)d_in[10];
  const float* rpb     = (const float*)d_in[11];
  const float* ln2_g   = (const float*)d_in[12];
  const float* ln2_b   = (const float*)d_in[13];
  const float* fc1_w   = (const float*)d_in[14];
  const float* fc1_b   = (const float*)d_in[15];
  const float* fc2_w   = (const float*)d_in[16];
  const float* fc2_b   = (const float*)d_in[17];
  const float* bn_n_g  = (const float*)d_in[18];
  const float* bn_n_b  = (const float*)d_in[19];
  const float* se_w1   = (const float*)d_in[20];
  const float* se_w2   = (const float*)d_in[21];
  const float* d1_w    = (const float*)d_in[22];
  const float* d1_b    = (const float*)d_in[23];
  const float* d1_bn_g = (const float*)d_in[24];
  const float* d1_bn_b = (const float*)d_in[25];
  const float* d2_w    = (const float*)d_in[26];
  const float* d2_b    = (const float*)d_in[27];
  const float* d2_bn_g = (const float*)d_in[28];
  const float* d2_bn_b = (const float*)d_in[29];
  (void)in_sizes; (void)n_in; (void)out_size; (void)ws_size;

  char* ws = (char*)d_ws;
  size_t off = 0;
  auto alloc = [&](size_t bytes) -> void* {
    void* p = ws + off;
    off += (bytes + 255) & ~(size_t)255;
    return p;
  };
  _Float16* xT      = (_Float16*)alloc((size_t)Mtok * 128 * 2);
  float*    xbuf    = (float*)   alloc((size_t)Mtok * 128 * 4);
  _Float16* h16     = (_Float16*)alloc((size_t)Mtok * 128 * 2);
  _Float16* qkvb    = (_Float16*)alloc((size_t)Mtok * 384 * 2);
  _Float16* aob     = (_Float16*)alloc((size_t)Mtok * 128 * 2);
  _Float16* mid     = (_Float16*)alloc((size_t)Mtok * 512 * 2);
  _Float16* xcT     = (_Float16*)alloc((size_t)Mtok * 512 * 2);
  _Float16* h1b     = (_Float16*)alloc((size_t)Mtok * 256 * 2);
  _Float16* qkvw16  = (_Float16*)alloc((size_t)4 * 6 * 384 * 128 * 2);
  _Float16* projw16 = (_Float16*)alloc((size_t)4 * 6 * 128 * 128 * 2);
  _Float16* fc1w16  = (_Float16*)alloc((size_t)4 * 6 * 512 * 128 * 2);
  _Float16* fc2w16  = (_Float16*)alloc((size_t)4 * 6 * 128 * 512 * 2);
  _Float16* convw16 = (_Float16*)alloc((size_t)4 * Kt * 128 * 128 * 2);
  _Float16* d1w16   = (_Float16*)alloc((size_t)Kt * 256 * 512 * 2);
  _Float16* d2w16   = (_Float16*)alloc((size_t)Kt * 128 * 256 * 2);
  float*    meanb   = (float*)alloc(Bn * 512 * 4);
  float*    t1b     = (float*)alloc(Bn * 256 * 4);
  float*    gateb   = (float*)alloc(Bn * 512 * 4);

  const int TPB = 256;
  auto cdiv = [](size_t a, size_t b) { return (int)((a + b - 1) / b); };

  /* weight conversion / packing */
  cvt_kernel<<<cdiv(4*6*384*128, TPB), TPB, 0, stream>>>(qkv_w, qkvw16, 4*6*384*128);
  cvt_kernel<<<cdiv(4*6*128*128, TPB), TPB, 0, stream>>>(proj_w, projw16, 4*6*128*128);
  cvt_kernel<<<cdiv(4*6*512*128, TPB), TPB, 0, stream>>>(fc1_w, fc1w16, 4*6*512*128);
  cvt_kernel<<<cdiv(4*6*128*512, TPB), TPB, 0, stream>>>(fc2_w, fc2w16, 4*6*128*512);
  packw_kernel<<<cdiv((size_t)4*128*128*Kt, TPB), TPB, 0, stream>>>(conv_w, convw16, 4, 128, 128);
  packw_kernel<<<cdiv((size_t)256*512*Kt, TPB), TPB, 0, stream>>>(d1_w, d1w16, 1, 256, 512);
  packw_kernel<<<cdiv((size_t)128*256*Kt, TPB), TPB, 0, stream>>>(d2_w, d2w16, 1, 128, 256);
  xpose_kernel<<<cdiv((size_t)Bn*CINn*Ln, TPB), TPB, 0, stream>>>(x, xT);

  const dim3 gB(128);

  for (int i = 0; i < 4; ++i) {
    /* branch dilated conv -> relu -> bn -> xbuf [B,L,C] fp32 */
    conv_gemm_kernel<<<dim3(128/64, Mtok/128), gB, 0, stream>>>(
        xT, convw16 + (size_t)i * Kt * 128 * 128, conv_b + i * 128,
        bn_a_g + i * 128, bn_a_b + i * 128, nullptr, xbuf, 128, 128, i + 1, CEPI_F32_BLC);
    for (int j = 0; j < 6; ++j) {
      const int sh = (j & 1) ? SHIFTn : 0;
      const size_t ij = (size_t)i * 6 + j;
      ln_kernel<<<Mtok/8, 256, 0, stream>>>(xbuf, ln1_g + ij*128, ln1_b + ij*128, h16, sh);
      gemm_f16_kernel<<<dim3(384/64, Mtok/128), gB, 0, stream>>>(
          h16, qkvw16 + ij * 384 * 128, qkv_b + ij * 384, qkvb, nullptr, 384, 128, EPI_F16, 0);
      attn_kernel<<<(Bn*NWn*NHn)/4, 128, 0, stream>>>(
          qkvb, rpb + ij * (2*WINn - 1) * NHn, aob, sh);
      gemm_f16_kernel<<<dim3(128/64, Mtok/128), gB, 0, stream>>>(
          aob, projw16 + ij * 128 * 128, proj_b + ij * 128, nullptr, xbuf, 128, 128, EPI_RES, sh);
      ln_kernel<<<Mtok/8, 256, 0, stream>>>(xbuf, ln2_g + ij*128, ln2_b + ij*128, h16, 0);
      gemm_f16_kernel<<<dim3(512/64, Mtok/128), gB, 0, stream>>>(
          h16, fc1w16 + ij * 512 * 128, fc1_b + ij * 512, mid, nullptr, 512, 128, EPI_GELU, 0);
      gemm_f16_kernel<<<dim3(128/64, Mtok/128), gB, 0, stream>>>(
          mid, fc2w16 + ij * 128 * 512, fc2_b + ij * 128, nullptr, xbuf, 128, 512, EPI_RES, 0);
    }
    branch_out_kernel<<<cdiv((size_t)Mtok * 128, TPB), TPB, 0, stream>>>(
        xbuf, bn_n_g + i * 128, bn_n_b + i * 128, xcT, i);
  }

  /* SE gating */
  mean_kernel<<<cdiv(Bn*512, TPB), TPB, 0, stream>>>(xcT, meanb);
  se1_kernel<<<cdiv(Bn*256, TPB), TPB, 0, stream>>>(meanb, se_w1, t1b);
  se2_kernel<<<cdiv(Bn*512, TPB), TPB, 0, stream>>>(t1b, se_w2, gateb);
  gate_kernel<<<cdiv((size_t)Mtok*512, TPB), TPB, 0, stream>>>(xcT, gateb);

  /* head convs */
  conv_gemm_kernel<<<dim3(256/64, Mtok/128), gB, 0, stream>>>(
      xcT, d1w16, d1_b, d1_bn_g, d1_bn_b, h1b, nullptr, 512, 256, 1, CEPI_F16_BLC);
  conv_gemm_kernel<<<dim3(128/64, Mtok/128), gB, 0, stream>>>(
      h1b, d2w16, d2_b, d2_bn_g, d2_bn_b, nullptr, (float*)d_out, 256, 128, 1, CEPI_F32_BCL);
}
